// AttendAndSpell_40114994544740
// MI455X (gfx1250) — matrix operand order repaired and checked by reading, compiled, not run on hardware
//
#include <hip/hip_runtime.h>
#include <math.h>

typedef __bf16 bf16;
typedef __bf16 v16bf __attribute__((ext_vector_type(16)));
typedef __bf16 v8bf  __attribute__((ext_vector_type(8)));
typedef float  v8f   __attribute__((ext_vector_type(8)));

#define H_   512
#define E_   256
#define B_   32
#define S_   256
#define L_   1024
#define G4H  2048            // 4*H
#define KZ   1280            // (E + H) + H = 768 + 512
#define KT_Z 40              // KZ / 32
#define NT_Z 128             // G4H / 16
#define KT_H 16              // 512 / 32
#define NT_H 32              // 512 / 16

union AF { v16bf v; v8bf h[2]; };

__device__ __forceinline__ v8f wmma_bf16(v16bf a, v16bf b, v8f c) {
  // D = A(16x32 bf16) * B(32x16 bf16) + C(16x16 f32)
  return __builtin_amdgcn_wmma_f32_16x16x32_bf16(false, a, false, b, (short)0, c,
                                                 false, false);
}

// ---------------- setup kernels ----------------

__global__ void k_conv_y(const float* __restrict__ y, bf16* __restrict__ y_bf) {
  int i = blockIdx.x * blockDim.x + threadIdx.x;
  if (i < B_ * S_ * E_) y_bf[i] = (bf16)y[i];
}

// Combined [W_ih | W_hh] -> WMMA-B tiled bf16 layout.
// tiled[((nt*KT + kt)*32 + lane)*16 + e] = W[n = nt*16 + (lane&15)][k = kt*32 + 8*(lane>>4) + (e&7) + 16*(e>>3)]
__global__ void k_fill_wz(const float* __restrict__ W_ih, const float* __restrict__ W_hh,
                          bf16* __restrict__ Wz_t) {
  int idx = blockIdx.x * blockDim.x + threadIdx.x;
  const int total = NT_Z * KT_Z * 32 * 16;
  if (idx >= total) return;
  int e = idx & 15;
  int t = idx >> 4;
  int lane = t & 31; t >>= 5;
  int kt = t % KT_Z;
  int nt = t / KT_Z;
  int n = nt * 16 + (lane & 15);
  int k = kt * 32 + ((lane >> 4) << 3) + (e & 7) + ((e >> 3) << 4);
  float v = (k < 768) ? W_ih[n * 768 + k] : W_hh[n * 512 + (k - 768)];
  Wz_t[idx] = (bf16)v;
}

// 512x512 weight (W_s or W_h) -> WMMA-B tiled bf16 layout
__global__ void k_fill_sq(const float* __restrict__ W, bf16* __restrict__ Wt) {
  int idx = blockIdx.x * blockDim.x + threadIdx.x;
  const int total = NT_H * KT_H * 32 * 16;
  if (idx >= total) return;
  int e = idx & 15;
  int t = idx >> 4;
  int lane = t & 31; t >>= 5;
  int kt = t & (KT_H - 1);
  int nt = t >> 4;
  int n = nt * 16 + (lane & 15);
  int k = kt * 32 + ((lane >> 4) << 3) + (e & 7) + ((e >> 3) << 4);
  Wt[idx] = (bf16)W[n * 512 + k];
}

__global__ void k_init(const float* __restrict__ b_ih, const float* __restrict__ b_hh,
                       float* __restrict__ bz, float* __restrict__ c_ws,
                       bf16* __restrict__ h_bf, bf16* __restrict__ att_bf) {
  int i = blockIdx.x * blockDim.x + threadIdx.x;
  if (i < G4H) bz[i] = b_ih[i] + b_hh[i];
  if (i < B_ * H_) {
    c_ws[i] = 0.f;
    h_bf[i] = (bf16)0.f;
    att_bf[i] = (bf16)0.f;
  }
}

// ---------------- h_e = encoder_h @ W_h^T + b_h  (32768 x 512, K=512), bf16 output ----------------

__global__ void k_he(const float* __restrict__ enc, const bf16* __restrict__ Wh_t,
                     const float* __restrict__ b_h, bf16* __restrict__ he) {
  const int lane = threadIdx.x & 31;
  const int half = lane >> 4, lm = lane & 15;
  int wave = blockIdx.x * (blockDim.x >> 5) + (threadIdx.x >> 5);
  const int nwave = gridDim.x * (blockDim.x >> 5);
  const int ntiles = (B_ * L_ / 16) * NT_H;   // 2048 * 32
  for (int tile = wave; tile < ntiles; tile += nwave) {
    int mt = tile >> 5;
    int nt = tile & 31;
    int row = mt * 16 + lm;                   // flat (b,l)
    const float* ap = enc + (size_t)row * H_;
    v8f acc = {0.f, 0.f, 0.f, 0.f, 0.f, 0.f, 0.f, 0.f};
    for (int kt = 0; kt < KT_H; ++kt) {
      int k0 = kt * 32 + half * 8;
      AF a;
#pragma unroll
      for (int e = 0; e < 8; ++e) {
        a.v[e]     = (bf16)ap[k0 + e];
        a.v[e + 8] = (bf16)ap[k0 + 16 + e];
      }
      v16bf bm = *(const v16bf*)(Wh_t + (((size_t)nt * KT_H + kt) * 32 + lane) * 16);
      acc = wmma_bf16(a.v, bm, acc);
    }
    int n0 = nt * 16 + lm;
    float bias = b_h[n0];
#pragma unroll
    for (int r = 0; r < 8; ++r) {
      int m = mt * 16 + r + half * 8;
      he[(size_t)m * H_ + n0] = (bf16)(acc[r] + bias);
    }
  }
}

// ---------------- per-step: gates = [y_t, att, h] @ [W_ih|W_hh]^T  (M=32, N=2048, K=1280) ----------------

__device__ __forceinline__ const v8bf* z_run(const bf16* __restrict__ ybrow,
                                             const bf16* __restrict__ att_bf,
                                             const bf16* __restrict__ h_bf,
                                             int b, int k0) {
  if (k0 < E_)       return (const v8bf*)(ybrow + k0);
  else if (k0 < 768) return (const v8bf*)(att_bf + b * H_ + (k0 - E_));
  else               return (const v8bf*)(h_bf + b * H_ + (k0 - 768));
}

__global__ void __launch_bounds__(128) k_gates(
    const bf16* __restrict__ y_bf, const bf16* __restrict__ att_bf,
    const bf16* __restrict__ h_bf, const bf16* __restrict__ Wz_t,
    float* __restrict__ gates, int s) {
  const int lane = threadIdx.x & 31;
  const int half = lane >> 4, lm = lane & 15;
  int wave = blockIdx.x * (blockDim.x >> 5) + (threadIdx.x >> 5);  // 0..255
  int mt = wave >> 7;
  int nt = wave & 127;
  int b = mt * 16 + lm;
  const bf16* ybrow = y_bf + ((size_t)b * S_ + s) * E_;
  v8f acc = {0.f, 0.f, 0.f, 0.f, 0.f, 0.f, 0.f, 0.f};
  for (int kt = 0; kt < KT_Z; ++kt) {
    int k0 = kt * 32 + half * 8;
    AF a;
    a.h[0] = *z_run(ybrow, att_bf, h_bf, b, k0);
    a.h[1] = *z_run(ybrow, att_bf, h_bf, b, k0 + 16);
    v16bf bm = *(const v16bf*)(Wz_t + (((size_t)nt * KT_Z + kt) * 32 + lane) * 16);
    acc = wmma_bf16(a.v, bm, acc);
  }
  int n0 = nt * 16 + lm;
#pragma unroll
  for (int r = 0; r < 8; ++r)
    gates[(mt * 16 + r + half * 8) * G4H + n0] = acc[r];
}

// ---------------- per-step: LSTM elementwise + s_e = h @ W_s^T + b_s (single WG) ----------------

__global__ void __launch_bounds__(1024) k_lstm_se(
    const float* __restrict__ gates, const float* __restrict__ bz,
    const float* __restrict__ b_s, const bf16* __restrict__ Ws_t,
    float* __restrict__ c_ws, bf16* __restrict__ h_bf,
    float* __restrict__ s_e, float* __restrict__ h_seq, int s) {
  __shared__ __align__(16) bf16 h_l[B_ * H_];   // 32 KB
  int tid = threadIdx.x;
  for (int idx = tid; idx < B_ * H_; idx += 1024) {
    int b = idx >> 9, j = idx & 511;
    const float* g = gates + b * G4H;
    float ig = g[j]        + bz[j];
    float fg = g[j + 512]  + bz[j + 512];
    float gg = g[j + 1024] + bz[j + 1024];
    float og = g[j + 1536] + bz[j + 1536];
    float si = 1.f / (1.f + __expf(-ig));
    float sf = 1.f / (1.f + __expf(-fg));
    float so = 1.f / (1.f + __expf(-og));
    float cn = sf * c_ws[idx] + si * tanhf(gg);
    float hn = so * tanhf(cn);
    c_ws[idx] = cn;
    bf16 hb = (bf16)hn;
    h_l[idx] = hb;
    h_bf[idx] = hb;
    h_seq[((size_t)b * S_ + s) * H_ + j] = hn;
  }
  __syncthreads();
  // s_e: M=32, N=512, K=512; 64 tiles over 32 waves
  const int lane = tid & 31, half = lane >> 4, lm = lane & 15;
  int wave = tid >> 5;
  for (int t = wave; t < 64; t += 32) {
    int mt = t >> 5, nt = t & 31;
    int b = mt * 16 + lm;
    const bf16* hp = h_l + b * H_;
    v8f acc = {0.f, 0.f, 0.f, 0.f, 0.f, 0.f, 0.f, 0.f};
    for (int kt = 0; kt < KT_H; ++kt) {
      int k0 = kt * 32 + half * 8;
      AF a;
      a.h[0] = *(const v8bf*)(hp + k0);
      a.h[1] = *(const v8bf*)(hp + k0 + 16);
      v16bf bm = *(const v16bf*)(Ws_t + (((size_t)nt * KT_H + kt) * 32 + lane) * 16);
      acc = wmma_bf16(a.v, bm, acc);
    }
    int n0 = nt * 16 + lm;
    float bias = b_s[n0];
#pragma unroll
    for (int r = 0; r < 8; ++r)
      s_e[(mt * 16 + r + half * 8) * H_ + n0] = acc[r] + bias;
  }
}

// ---------------- per-step: e = s_e . h_e, softmax, att = a . h_e  (one WG per batch) ----------------

__global__ void __launch_bounds__(256) k_attn(
    const bf16* __restrict__ he, const float* __restrict__ s_e,
    float* __restrict__ att_seq, bf16* __restrict__ att_bf, int s) {
  __shared__ float se_l[H_];
  __shared__ float e_l[L_];
  __shared__ float red[256];
  const int tid = threadIdx.x, b = blockIdx.x;
  const int lane = tid & 31, wave = tid >> 5;   // 8 waves
  for (int i = tid; i < H_; i += 256) se_l[i] = s_e[b * H_ + i];
  __syncthreads();
  const bf16* heb = he + (size_t)b * L_ * H_;
  for (int l = wave; l < L_; l += 8) {
    const bf16* row = heb + (size_t)l * H_;
    float acc = 0.f;
#pragma unroll
    for (int j = 0; j < 16; ++j) {
      int hh = lane + 32 * j;
      acc += (float)row[hh] * se_l[hh];
    }
#pragma unroll
    for (int off = 16; off > 0; off >>= 1)
      acc += __shfl_down(acc, off, 32);
    if (lane == 0) e_l[l] = acc;
  }
  __syncthreads();
  // softmax over L
  float m = -3.4e38f;
  for (int l = tid; l < L_; l += 256) m = fmaxf(m, e_l[l]);
  red[tid] = m; __syncthreads();
  for (int off = 128; off > 0; off >>= 1) {
    if (tid < off) red[tid] = fmaxf(red[tid], red[tid + off]);
    __syncthreads();
  }
  float mx = red[0];
  __syncthreads();
  float ssum = 0.f;
  for (int l = tid; l < L_; l += 256) {
    float ex = __expf(e_l[l] - mx);
    e_l[l] = ex;
    ssum += ex;
  }
  red[tid] = ssum; __syncthreads();
  for (int off = 128; off > 0; off >>= 1) {
    if (tid < off) red[tid] += red[tid + off];
    __syncthreads();
  }
  float inv = 1.f / red[0];
  for (int l = tid; l < L_; l += 256) e_l[l] *= inv;
  __syncthreads();
  // att[b,h] = sum_l a[l] * he[b,l,h]; thread handles h = tid and tid+256
  float a0 = 0.f, a1 = 0.f;
  for (int l = 0; l < L_; ++l) {
    float al = e_l[l];
    const bf16* row = heb + (size_t)l * H_;
    a0 += al * (float)row[tid];
    a1 += al * (float)row[tid + 256];
  }
  size_t o = ((size_t)b * S_ + s) * H_;
  att_seq[o + tid] = a0;
  att_seq[o + tid + 256] = a1;
  att_bf[b * H_ + tid] = (bf16)a0;
  att_bf[b * H_ + tid + 256] = (bf16)a1;
}

// ---------------- launch ----------------

extern "C" void kernel_launch(void* const* d_in, const int* in_sizes, int n_in,
                              void* d_out, int out_size, void* d_ws, size_t ws_size,
                              hipStream_t stream) {
  (void)in_sizes; (void)n_in; (void)out_size; (void)ws_size;
  const float* y    = (const float*)d_in[0];
  const float* enc  = (const float*)d_in[1];
  const float* W_ih = (const float*)d_in[2];
  const float* b_ih = (const float*)d_in[3];
  const float* W_hh = (const float*)d_in[4];
  const float* b_hh = (const float*)d_in[5];
  const float* W_s  = (const float*)d_in[6];
  const float* b_s  = (const float*)d_in[7];
  const float* W_h  = (const float*)d_in[8];
  const float* b_h  = (const float*)d_in[9];

  float* h_seq   = (float*)d_out;
  float* att_seq = h_seq + (size_t)B_ * S_ * H_;

  char* p = (char*)d_ws;
  auto carve = [&](size_t bytes) -> void* {
    void* r = (void*)p;
    p += (bytes + 255) & ~(size_t)255;
    return r;
  };
  bf16*  he     = (bf16*)carve((size_t)B_ * L_ * H_ * 2);       // 32 MB
  bf16*  Wz_t   = (bf16*)carve((size_t)NT_Z * KT_Z * 512 * 2);  // 5 MB
  bf16*  Ws_t   = (bf16*)carve((size_t)NT_H * KT_H * 512 * 2);
  bf16*  Wh_t   = (bf16*)carve((size_t)NT_H * KT_H * 512 * 2);
  bf16*  y_bf   = (bf16*)carve((size_t)B_ * S_ * E_ * 2);       // 4 MB
  float* bz     = (float*)carve((size_t)G4H * 4);
  float* gates  = (float*)carve((size_t)B_ * G4H * 4);
  float* c_ws   = (float*)carve((size_t)B_ * H_ * 4);
  float* s_e    = (float*)carve((size_t)B_ * H_ * 4);
  bf16*  h_bf   = (bf16*)carve((size_t)B_ * H_ * 2);
  bf16*  att_bf = (bf16*)carve((size_t)B_ * H_ * 2);

  // one-time setup
  k_conv_y<<<(B_ * S_ * E_ + 255) / 256, 256, 0, stream>>>(y, y_bf);
  k_fill_wz<<<(NT_Z * KT_Z * 512 + 255) / 256, 256, 0, stream>>>(W_ih, W_hh, Wz_t);
  k_fill_sq<<<(NT_H * KT_H * 512 + 255) / 256, 256, 0, stream>>>(W_s, Ws_t);
  k_fill_sq<<<(NT_H * KT_H * 512 + 255) / 256, 256, 0, stream>>>(W_h, Wh_t);
  k_init<<<(B_ * H_ + 255) / 256, 256, 0, stream>>>(b_ih, b_hh, bz, c_ws, h_bf, att_bf);
  k_he<<<2048, 256, 0, stream>>>(enc, Wh_t, b_h, he);

  // sequential decoder scan
  for (int s = 0; s < S_; ++s) {
    k_gates<<<64, 128, 0, stream>>>(y_bf, att_bf, h_bf, Wz_t, gates, s);
    k_lstm_se<<<1, 1024, 0, stream>>>(gates, bz, b_s, Ws_t, c_ws, h_bf, s_e, h_seq, s);
    k_attn<<<32, 256, 0, stream>>>(he, s_e, att_seq, att_bf, s);
  }
}
